// Gated_Bottleneck_Convolution_TV_37331855737299
// MI455X (gfx1250) — matrix operand order repaired and checked by reading, not compiled, not run on hardware
//
#include <hip/hip_runtime.h>

// ---------------------------------------------------------------------------
// Gated Bottleneck Convolution (TVConv) for MI455X / gfx1250.
// Heavy FLOPs (weight_gen convs, ~15.8 GFLOP) run as im2col GEMMs on the
// WMMA pipe: v_wmma_f32_16x16x32_f16 (f16 in, f32 accumulate), with the
// block-shared B tile staged into LDS via async global->LDS copies
// (double-buffered, ASYNCcnt-synchronized). Everything else (LN/GN
// reductions, tvconv apply, gating, residual) is bandwidth-bound and fused
// into simple VALU kernels.
// ---------------------------------------------------------------------------

typedef _Float16 half_t;
typedef __attribute__((ext_vector_type(16))) _Float16 v16h;
typedef __attribute__((ext_vector_type(8)))  _Float16 v8h;
typedef __attribute__((ext_vector_type(8)))  float    v8f;
typedef __attribute__((ext_vector_type(4)))  int      v4i;

#define HWN   4096          // 64*64 spatial
#define CCH   128           // channels
#define BBN   4             // batch
#define PADW  66
#define PADCH (PADW*PADW)   // 4356, padded per-channel plane
#define EPSF  1e-5f

#if defined(__AMDGCN__) && __has_builtin(__builtin_amdgcn_global_load_async_to_lds_b128) && __has_builtin(__builtin_amdgcn_s_wait_asynccnt)
#define USE_ASYNC_LDS 1
typedef __attribute__((address_space(1))) v4i* gv4i_p;   // global b128
typedef __attribute__((address_space(3))) v4i* lv4i_p;   // LDS b128
#else
#define USE_ASYNC_LDS 0
#endif

// ---------------- WMMA GEMM:  O[M][4096] = W[M][Kpad] x Col[4096][Kpad]^T ---
// Col is stored N-major. Block = 4 waves, tile 64(M) x 64(N).
// The 32(K) x 64(N) B tile (4 KB) is shared by all 4 waves: staged into LDS
// (async, double buffered). Each wave: 1 A fragment + 4 WMMAs per K-step.
__global__ __launch_bounds__(128)
void k_gemm_wmma(const half_t* __restrict__ Wm, const half_t* __restrict__ Cm,
                 float* __restrict__ O, int M, int Kpad) {
  __shared__ half_t lbs[2][64 * 32];   // [buf][n_local * 32 + k]
  const int t    = threadIdx.x;
  const int lane = t & 31;
  const int wave = t >> 5;
  const int hi   = lane >> 4;
  const int lo   = lane & 15;
  const int m0 = blockIdx.y * 64 + wave * 16;
  const int n0 = blockIdx.x * 64;

  // Stage one 64x32 B tile into lbs[buf]: 256 16-byte chunks, 2 per thread.
  // Chunk c -> n_local = c>>2, k-halves = (c&3)*8, LDS offset = c*16 bytes.
  auto stage = [&](int buf, int k0) {
#pragma unroll
    for (int c0 = 0; c0 < 256; c0 += 128) {
      int c = c0 + t;
      const half_t* g = Cm + (size_t)(n0 + (c >> 2)) * Kpad + k0 + (c & 3) * 8;
      half_t* l = &lbs[buf][0] + c * 8;
#if USE_ASYNC_LDS
      __builtin_amdgcn_global_load_async_to_lds_b128(
          (gv4i_p)(uintptr_t)g, (lv4i_p)(uintptr_t)l, 0, 0);
#else
      *(v8h*)l = *(const v8h*)g;
#endif
    }
  };

  const int steps = Kpad >> 5;
  stage(0, 0);

  v8f acc[4] = {v8f{}, v8f{}, v8f{}, v8f{}};
  for (int s = 0; s < steps; ++s) {
    const int k0 = s << 5;
#if USE_ASYNC_LDS
    __builtin_amdgcn_s_wait_asynccnt(0);
#endif
    __syncthreads();
    if (s + 1 < steps) stage((s + 1) & 1, k0 + 32);

    // A fragment (16x32 f16): lanes 0-15 -> K {k0..k0+7, k0+16..k0+23},
    // lanes 16-31 -> K {k0+8..k0+15, k0+24..k0+31}, row = m0+lo.
    const half_t* ap = Wm + (size_t)(m0 + lo) * Kpad + k0 + hi * 8;
    v8h a_lo = *(const v8h*)(ap);
    v8h a_hi = *(const v8h*)(ap + 16);
    v16h a;
#pragma unroll
    for (int i = 0; i < 8; ++i) { a[i] = a_lo[i]; a[8 + i] = a_hi[i]; }

    const half_t* lb = &lbs[s & 1][0];
#pragma unroll
    for (int sub = 0; sub < 4; ++sub) {
      // B fragment (32x16): lane column = sub*16 + lo, K {hi*8..hi*8+7,
      // 16+hi*8..16+hi*8+7} within the tile row (matches A).
      const half_t* bp = lb + (size_t)((sub * 16 + lo) * 32 + hi * 8);
      v8h b_lo = *(const v8h*)(bp);
      v8h b_hi = *(const v8h*)(bp + 16);
      v16h b;
#pragma unroll
      for (int i = 0; i < 8; ++i) { b[i] = b_lo[i]; b[8 + i] = b_hi[i]; }
      acc[sub] = __builtin_amdgcn_wmma_f32_16x16x32_f16(
          false, a, false, b, (short)0, acc[sub], false, false);
    }
  }

  // C/D layout: VGPR r, lane l -> M = m0 + r + 8*(l>>4), N = n_tile + (l&15).
#pragma unroll
  for (int sub = 0; sub < 4; ++sub) {
#pragma unroll
    for (int r = 0; r < 8; ++r) {
      int row = m0 + r + 8 * hi;
      O[(size_t)row * HWN + n0 + sub * 16 + lo] = acc[sub][r];
    }
  }
}

// ---------------- weight f32 -> f16, K padded to Kpad with zeros ------------
__global__ void k_cvt_w(const float* __restrict__ w, int K,
                        half_t* __restrict__ wf, int Kpad, int total) {
  int idx = blockIdx.x * blockDim.x + threadIdx.x;
  if (idx >= total) return;
  int m = idx / Kpad, kk = idx - m * Kpad;
  wf[idx] = (kk < K) ? (half_t)w[(size_t)m * K + kk] : (half_t)0.0f;
}

// ---------------- raw f32 act -> padded f16 planes (layer-0 input) ----------
__global__ void k_pack_padact(const float* __restrict__ src,
                              half_t* __restrict__ dst, int total) {
  int idx = blockIdx.x * blockDim.x + threadIdx.x;
  if (idx >= total) return;
  int c = idx >> 12, n = idx & 4095, h = n >> 6, w = n & 63;
  dst[(size_t)c * PADCH + (h + 1) * PADW + (w + 1)] = (half_t)src[idx];
}

// ---------------- sum / sumsq reduction (for LayerNorm) ---------------------
__global__ void k_reduce2(const float* __restrict__ y, float* __restrict__ st) {
  __shared__ float s0[256], s1[256];
  int t = threadIdx.x;
  float a = 0.f, b = 0.f;
#pragma unroll
  for (int j = 0; j < 4; ++j) {
    float v = y[blockIdx.x * 256 + t + j * 65536];
    a += v; b += v * v;
  }
  s0[t] = a; s1[t] = b;
  __syncthreads();
  for (int off = 128; off > 0; off >>= 1) {
    if (t < off) { s0[t] += s0[t + off]; s1[t] += s1[t + off]; }
    __syncthreads();
  }
  if (t == 0) { atomicAdd(&st[0], s0[0]); atomicAdd(&st[1], s1[0]); }
}

// ---------------- LayerNorm + ReLU -> padded f16 act ------------------------
__global__ void k_ln_relu_pad(const float* __restrict__ y,
                              const float* __restrict__ st,
                              const float* __restrict__ g,
                              const float* __restrict__ b,
                              half_t* __restrict__ dst) {
  int idx = blockIdx.x * blockDim.x + threadIdx.x;  // 262144 threads
  const float inv_n = 1.0f / 262144.0f;
  float mu = st[0] * inv_n;
  float var = st[1] * inv_n - mu * mu;
  float r = rsqrtf(var + EPSF);
  float v = (y[idx] - mu) * r * g[idx] + b[idx];
  v = fmaxf(v, 0.0f);
  int c = idx >> 12, n = idx & 4095, h = n >> 6, w = n & 63;
  dst[(size_t)c * PADCH + (h + 1) * PADW + (w + 1)] = (half_t)v;
}

// ---------------- im2col (3x3, pad 1) into N-major col[n][Kpad] -------------
__global__ void k_im2col(const half_t* __restrict__ pad,
                         half_t* __restrict__ col, int Cin, int Kpad, int total) {
  int idx = blockIdx.x * blockDim.x + threadIdx.x;  // N * Cin threads
  if (idx >= total) return;
  int ci = idx & (Cin - 1);
  int n  = idx / Cin;
  int h = n >> 6, w = n & 63;
  half_t* dst = col + (size_t)n * Kpad + ci * 9;
  const half_t* src = pad + (size_t)ci * PADCH + h * PADW + w;
#pragma unroll
  for (int dh = 0; dh < 3; ++dh)
#pragma unroll
    for (int dw = 0; dw < 3; ++dw)
      dst[dh * 3 + dw] = src[dh * PADW + dw];
}

// ---------------- TVConv apply (k=3): per-pixel 3x3 weights -----------------
__global__ void k_tvconv3(const float* __restrict__ x,
                          const float* __restrict__ wgt,
                          float* __restrict__ out) {
  int idx = blockIdx.x * blockDim.x + threadIdx.x;  // B*C*4096
  int b = idx >> 19, c = (idx >> 12) & 127, n = idx & 4095;
  int h = n >> 6, w = n & 63;
  const float* xp = x + ((size_t)(b * CCH + c) << 12);
  float acc = 0.f;
#pragma unroll
  for (int t = 0; t < 9; ++t) {
    int i = t / 3, j = t - 3 * i;
    int hh = h + i - 1, ww = w + j - 1;
    float xv = (hh >= 0 && hh < 64 && ww >= 0 && ww < 64)
                   ? xp[(hh << 6) + ww] : 0.f;
    acc = fmaf(wgt[((size_t)(c * 9 + t) << 12) + n], xv, acc);
  }
  out[idx] = acc;
}

// ---------------- TVConv apply (k=1) ----------------------------------------
__global__ void k_tvconv1(const float* __restrict__ x,
                          const float* __restrict__ wgt,
                          float* __restrict__ out) {
  int idx = blockIdx.x * blockDim.x + threadIdx.x;  // B*C*4096
  int c = (idx >> 12) & 127, n = idx & 4095;
  out[idx] = wgt[((size_t)c << 12) + n] * x[idx];
}

// ---------------- GroupNorm stats: one block per (batch, group) -------------
__global__ void k_gn_stats(const float* __restrict__ x, float* __restrict__ st) {
  __shared__ float s0[256], s1[256];
  int t = threadIdx.x;
  const float* p = x + ((size_t)blockIdx.x << 16);  // 16 ch * 4096
  float a = 0.f, b = 0.f;
  for (int i = t; i < 65536; i += 256) {
    float v = p[i];
    a += v; b += v * v;
  }
  s0[t] = a; s1[t] = b;
  __syncthreads();
  for (int off = 128; off > 0; off >>= 1) {
    if (t < off) { s0[t] += s0[t + off]; s1[t] += s1[t + off]; }
    __syncthreads();
  }
  if (t == 0) {
    float mu = s0[0] / 65536.0f;
    st[2 * blockIdx.x]     = mu;
    st[2 * blockIdx.x + 1] = s1[0] / 65536.0f - mu * mu;
  }
}

// ---------------- GroupNorm apply + ReLU (+ optional gate / residual) -------
__global__ void k_gn_apply(const float* __restrict__ x,
                           const float* __restrict__ st,
                           const float* __restrict__ gamma,
                           const float* __restrict__ beta,
                           const float* __restrict__ mul,   // may be null
                           const float* __restrict__ add,   // may be null
                           float* __restrict__ out) {
  int idx = blockIdx.x * blockDim.x + threadIdx.x;  // B*C*4096
  int c = (idx >> 12) & 127;
  int sidx = idx >> 16;  // (b*8 + group)
  float mu = st[2 * sidx], var = st[2 * sidx + 1];
  float v = (x[idx] - mu) * rsqrtf(var + EPSF) * gamma[c] + beta[c];
  v = fmaxf(v, 0.0f);
  if (mul) v *= mul[idx];
  if (add) v += add[idx];
  out[idx] = v;
}

// ===========================================================================
// Host side
// ===========================================================================
struct Branch {
  const float* posi;
  const float* conv[4];
  const float* g[4];
  const float* b[4];
  const float* wout;
};

static void get_branch(void* const* d_in, int base, Branch* br) {
  br->posi = (const float*)d_in[base];
  for (int i = 0; i < 4; ++i) {
    br->conv[i] = (const float*)d_in[base + 1 + i];
    br->g[i]    = (const float*)d_in[base + 5 + i];
    br->b[i]    = (const float*)d_in[base + 9 + i];
  }
  br->wout = (const float*)d_in[base + 13];
}

// Run one weight_gen CNN, producing wgt (f32, [k2C][4096]).
static void weight_gen(const Branch& br, int k2C,
                       half_t* padact, half_t* col, half_t* wf16,
                       float* ybuf, float* stats, float* wgt,
                       hipStream_t s) {
  const int KP0 = 64, KP = 576;
  (void)hipMemsetAsync(padact, 0, (size_t)64 * PADCH * sizeof(half_t), s);
  (void)hipMemsetAsync(col, 0, (size_t)HWN * KP * sizeof(half_t), s);

  // layer 0: conv 4 -> 64 (K = 36 padded to 64)
  k_pack_padact<<<64, 256, 0, s>>>(br.posi, padact, 4 * HWN);
  k_im2col<<<64, 256, 0, s>>>(padact, col, 4, KP0, HWN * 4);
  k_cvt_w<<<(64 * KP0) / 256, 256, 0, s>>>(br.conv[0], 36, wf16, KP0, 64 * KP0);
  k_gemm_wmma<<<dim3(HWN / 64, 1), 128, 0, s>>>(wf16, col, ybuf, 64, KP0);

  for (int i = 0; i < 4; ++i) {
    (void)hipMemsetAsync(stats, 0, 2 * sizeof(float), s);
    k_reduce2<<<256, 256, 0, s>>>(ybuf, stats);
    k_ln_relu_pad<<<1024, 256, 0, s>>>(ybuf, stats, br.g[i], br.b[i], padact);
    k_im2col<<<1024, 256, 0, s>>>(padact, col, 64, KP, HWN * 64);
    if (i < 3) {
      k_cvt_w<<<(64 * KP) / 256, 256, 0, s>>>(br.conv[i + 1], KP, wf16, KP, 64 * KP);
      k_gemm_wmma<<<dim3(HWN / 64, 1), 128, 0, s>>>(wf16, col, ybuf, 64, KP);
    }
  }
  // output conv: 64 -> k2C
  k_cvt_w<<<(k2C * KP + 255) / 256, 256, 0, s>>>(br.wout, KP, wf16, KP, k2C * KP);
  k_gemm_wmma<<<dim3(HWN / 64, k2C / 64), 128, 0, s>>>(wf16, col, wgt, k2C, KP);
}

extern "C" void kernel_launch(void* const* d_in, const int* in_sizes, int n_in,
                              void* d_out, int out_size, void* d_ws, size_t ws_size,
                              hipStream_t stream) {
  (void)in_sizes; (void)n_in; (void)out_size; (void)ws_size;
  const float* x = (const float*)d_in[0];
  Branch p1, p2, p3, p4;
  get_branch(d_in, 1, &p1);
  get_branch(d_in, 15, &p2);
  get_branch(d_in, 29, &p3);
  get_branch(d_in, 43, &p4);
  const float* gn_g[4] = { (const float*)d_in[57], (const float*)d_in[59],
                           (const float*)d_in[61], (const float*)d_in[63] };
  const float* gn_b[4] = { (const float*)d_in[58], (const float*)d_in[60],
                           (const float*)d_in[62], (const float*)d_in[64] };
  float* out = (float*)d_out;

  // ---- workspace carve-up (256B aligned) ----
  char* ws = (char*)d_ws;
  size_t ofs = 0;
  auto alloc = [&](size_t bytes) {
    void* p = ws + ofs;
    ofs = (ofs + bytes + 255) & ~(size_t)255;
    return p;
  };
  half_t* padact = (half_t*)alloc((size_t)64 * PADCH * sizeof(half_t));   // 0.56 MB
  half_t* col    = (half_t*)alloc((size_t)HWN * 576 * sizeof(half_t));    // 4.7 MB
  half_t* wf16   = (half_t*)alloc((size_t)1152 * 576 * sizeof(half_t));   // 1.3 MB
  float*  ybuf   = (float*)alloc((size_t)64 * HWN * sizeof(float));       // 1 MB
  float*  wgt    = (float*)alloc((size_t)1152 * HWN * sizeof(float));     // 18.9 MB
  float*  tbuf   = (float*)alloc((size_t)BBN * CCH * HWN * sizeof(float));// 8.4 MB
  float*  h1buf  = (float*)alloc((size_t)BBN * CCH * HWN * sizeof(float));// 8.4 MB
  float*  hbuf   = (float*)alloc((size_t)BBN * CCH * HWN * sizeof(float));// 8.4 MB
  float*  stats  = (float*)alloc(256);
  float*  gnst   = (float*)alloc(256);

  const int ELEMS = BBN * CCH * HWN;      // 2,097,152
  const int EB = ELEMS / 256;             // 8192 blocks

  // branch 1: h1 = relu(gn1(tvconv3(x, wgt1)))
  weight_gen(p1, 9 * CCH, padact, col, wf16, ybuf, stats, wgt, stream);
  k_tvconv3<<<EB, 256, 0, stream>>>(x, wgt, tbuf);
  k_gn_stats<<<32, 256, 0, stream>>>(tbuf, gnst);
  k_gn_apply<<<EB, 256, 0, stream>>>(tbuf, gnst, gn_g[0], gn_b[0],
                                     nullptr, nullptr, h1buf);

  // branch 2: h1b = relu(gn2(tvconv3(h1, wgt2)))
  weight_gen(p2, 9 * CCH, padact, col, wf16, ybuf, stats, wgt, stream);
  k_tvconv3<<<EB, 256, 0, stream>>>(h1buf, wgt, tbuf);
  k_gn_stats<<<32, 256, 0, stream>>>(tbuf, gnst);
  k_gn_apply<<<EB, 256, 0, stream>>>(tbuf, gnst, gn_g[1], gn_b[1],
                                     nullptr, nullptr, h1buf);

  // branch 3 (gate): h = relu(gn3(tvconv1(x, wgt3))) * h1b
  weight_gen(p3, CCH, padact, col, wf16, ybuf, stats, wgt, stream);
  k_tvconv1<<<EB, 256, 0, stream>>>(x, wgt, tbuf);
  k_gn_stats<<<32, 256, 0, stream>>>(tbuf, gnst);
  k_gn_apply<<<EB, 256, 0, stream>>>(tbuf, gnst, gn_g[2], gn_b[2],
                                     h1buf, nullptr, hbuf);

  // branch 4: out = relu(gn4(tvconv1(h, wgt4))) + x
  weight_gen(p4, CCH, padact, col, wf16, ybuf, stats, wgt, stream);
  k_tvconv1<<<EB, 256, 0, stream>>>(hbuf, wgt, tbuf);
  k_gn_stats<<<32, 256, 0, stream>>>(tbuf, gnst);
  k_gn_apply<<<EB, 256, 0, stream>>>(tbuf, gnst, gn_g[3], gn_b[3],
                                     nullptr, x, out);
}
